// LinearReadoutImplicitModule_74792560493198
// MI455X (gfx1250) — matrix operand order repaired and checked by reading, not compiled, not run on hardware
//
#include <hip/hip_runtime.h>

// ---------------------------------------------------------------------------
// Problem constants (from reference)
// ---------------------------------------------------------------------------
#define KBLK   8
#define BLK    1024
#define TOTAL  8192          // KBLK * BLK
#define BATCH  4096

// GEMM tiling: 128x128 tile, K-step 32 (bf16), 256 threads = 8 wave32s.
#define BM     128
#define BN     128
#define BKT    32
#define BKP    40            // padded LDS row stride (shorts); 16B-aligned chunks, spreads banks
#define KTILES (BLK / BKT)   // 32
#define NBUF   3             // triple-buffered LDS stages

typedef __attribute__((ext_vector_type(16))) __bf16 v16bf;
typedef __attribute__((ext_vector_type(8)))  float  v8f;

#define GLOBAL_AS __attribute__((address_space(1)))
#define LDS_AS    __attribute__((address_space(3)))

// Matches the builtin's parameter pointee type exactly (GCC vector_size style).
typedef int b128_t __attribute__((vector_size(16)));

// ---------------------------------------------------------------------------
// CDNA5 async global->LDS copy (ASYNCcnt-tracked, no VGPR staging)
// ---------------------------------------------------------------------------
__device__ __forceinline__ void async_copy_b128(const unsigned short* gsrc,
                                                unsigned short* ldst) {
#if __has_builtin(__builtin_amdgcn_global_load_async_to_lds_b128)
    __builtin_amdgcn_global_load_async_to_lds_b128(
        (GLOBAL_AS b128_t*)gsrc, (LDS_AS b128_t*)ldst, /*offset=*/0, /*cpol=*/0);
#else
    unsigned laddr = (unsigned)(unsigned long long)(LDS_AS void*)ldst;
    asm volatile("global_load_async_to_lds_b128 %0, %1, off"
                 :: "v"(laddr), "v"(gsrc) : "memory");
#endif
}

template <int N>
__device__ __forceinline__ void wait_asynccnt() {
#if __has_builtin(__builtin_amdgcn_s_wait_asynccnt)
    __builtin_amdgcn_s_wait_asynccnt(N);
#else
    asm volatile("s_wait_asynccnt %0" :: "n"(N) : "memory");
#endif
}

__device__ __forceinline__ unsigned short f2bf(float f) {
    // round-to-nearest-even float -> bf16 (raw bits)
    unsigned int u = __float_as_uint(f);
    unsigned int r = 0x7FFFu + ((u >> 16) & 1u);
    return (unsigned short)((u + r) >> 16);
}

// ---------------------------------------------------------------------------
// Prep kernels: H = bf16(tanh(v)), Wb = bf16(W).  8 elements / thread.
// ---------------------------------------------------------------------------
__global__ void tanh_to_bf16(const float* __restrict__ src,
                             unsigned short* __restrict__ dst) {
    long long i = (long long)blockIdx.x * blockDim.x + threadIdx.x;
    const float4* s4 = (const float4*)src;
    float4 a = s4[2 * i];
    float4 c = s4[2 * i + 1];
    uint4 o;
    o.x = (unsigned)f2bf(tanhf(a.x)) | ((unsigned)f2bf(tanhf(a.y)) << 16);
    o.y = (unsigned)f2bf(tanhf(a.z)) | ((unsigned)f2bf(tanhf(a.w)) << 16);
    o.z = (unsigned)f2bf(tanhf(c.x)) | ((unsigned)f2bf(tanhf(c.y)) << 16);
    o.w = (unsigned)f2bf(tanhf(c.z)) | ((unsigned)f2bf(tanhf(c.w)) << 16);
    ((uint4*)dst)[i] = o;
}

__global__ void conv_to_bf16(const float* __restrict__ src,
                             unsigned short* __restrict__ dst) {
    long long i = (long long)blockIdx.x * blockDim.x + threadIdx.x;
    const float4* s4 = (const float4*)src;
    float4 a = s4[2 * i];
    float4 c = s4[2 * i + 1];
    uint4 o;
    o.x = (unsigned)f2bf(a.x) | ((unsigned)f2bf(a.y) << 16);
    o.y = (unsigned)f2bf(a.z) | ((unsigned)f2bf(a.w) << 16);
    o.z = (unsigned)f2bf(c.x) | ((unsigned)f2bf(c.y) << 16);
    o.w = (unsigned)f2bf(c.z) | ((unsigned)f2bf(c.w) << 16);
    ((uint4*)dst)[i] = o;
}

// ---------------------------------------------------------------------------
// Block GEMM: for block k, Y = tanh(V_k) @ W_k^T ; out[:, (k+1)%8 block] =
// x + bias_k + Y.  A = H rows, B = W rows (row-major == WMMA K x N layout).
// Global -> LDS via async-DMA (no VGPR staging), triple buffered.
// ---------------------------------------------------------------------------
__global__ __launch_bounds__(256) void block_gemm(
    const unsigned short* __restrict__ H,    // [BATCH][TOTAL] bf16(tanh(v))
    const unsigned short* __restrict__ Wb,   // [KBLK][BLK][BLK] bf16(W)
    const float* __restrict__ x,             // [BATCH][TOTAL]
    const float* __restrict__ bias,          // [KBLK][BLK]
    float* __restrict__ out)                 // [BATCH][TOTAL]
{
    __shared__ unsigned short As[NBUF][BM][BKP];
    __shared__ unsigned short Bs[NBUF][BN][BKP];

    const int k   = blockIdx.z;              // which diagonal block
    const int mt0 = blockIdx.y * BM;         // batch-row base
    const int nt0 = blockIdx.x * BN;         // out-col base (within block)

    const int tid   = threadIdx.x;
    const int lane  = tid & 31;
    const int wave  = tid >> 5;
    const int waveM = wave >> 1;             // 0..3 : 32 rows each
    const int waveN = wave & 1;              // 0..1 : 64 cols each

    // async-copy mapping: thread -> 16B chunk; 4 chunks/thread covers A+B tile
    const int arow = tid >> 2;               // 0..63
    const int acol = (tid & 3) * 8;          // 0,8,16,24 (shorts)

    const unsigned short* gA0 =
        H + (long long)(mt0 + arow) * TOTAL + (long long)k * BLK + acol;
    const unsigned short* gA1 = gA0 + 64LL * TOTAL;
    const unsigned short* gB0 =
        Wb + (long long)k * BLK * BLK + (long long)(nt0 + arow) * BLK + acol;
    const unsigned short* gB1 = gB0 + 64LL * BLK;

    auto async_tile = [&](int buf, int kt) {
        const int ko = kt * BKT;
        async_copy_b128(gA0 + ko, &As[buf][arow][acol]);
        async_copy_b128(gA1 + ko, &As[buf][arow + 64][acol]);
        async_copy_b128(gB0 + ko, &Bs[buf][arow][acol]);
        async_copy_b128(gB1 + ko, &Bs[buf][arow + 64][acol]);
    };

    // A fragment (16x32 bf16): lanes 0-15 row m K{0-7,16-23}, lanes 16-31 K{8-15,24-31}
    auto loadA = [&](int buf, int mt) -> v16bf {
        int row = waveM * 32 + mt * 16 + (lane & 15);
        int kh  = lane >> 4;
        const unsigned short* p = &As[buf][row][0];
        union { uint4 u[2]; v16bf v; } t;
        t.u[0] = *(const uint4*)(p + kh * 8);
        t.u[1] = *(const uint4*)(p + kh * 8 + 16);
        return t.v;
    };
    // B fragment (32x16 bf16): lane = column; lanes 0-15 K{0-7,16-23}, lanes 16-31 K{8-15,24-31}
    auto loadB = [&](int buf, int nt) -> v16bf {
        int col = waveN * 64 + nt * 16 + (lane & 15);
        int kh  = lane >> 4;
        const unsigned short* p = &Bs[buf][col][0];
        union { uint4 u[2]; v16bf v; } t;
        t.u[0] = *(const uint4*)(p + kh * 8);
        t.u[1] = *(const uint4*)(p + kh * 8 + 16);
        return t.v;
    };

    v8f acc[2][4];
#pragma unroll
    for (int mt = 0; mt < 2; ++mt)
#pragma unroll
        for (int nt = 0; nt < 4; ++nt)
            acc[mt][nt] = (v8f){0.f, 0.f, 0.f, 0.f, 0.f, 0.f, 0.f, 0.f};

    // Prologue: two tiles in flight (8 async ops, 4 per tile).
    async_tile(0, 0);
    async_tile(1, 1);

    for (int kt = 0; kt < KTILES; ++kt) {
        // Oldest in-flight tile (kt) complete: async loads finish in order,
        // so <=4 outstanding leaves only tile kt+1 pending.
        if (kt + 1 < KTILES) wait_asynccnt<4>(); else wait_asynccnt<0>();
        // All waves have tile kt resident AND finished reading buf (kt+2)%NBUF
        // (last read at iteration kt-1), so it is safe to refill it below.
        __syncthreads();
        if (kt + 2 < KTILES) async_tile((kt + 2) % NBUF, kt + 2);

        const int cur = kt % NBUF;
        v16bf af[2], bf[4];
#pragma unroll
        for (int mt = 0; mt < 2; ++mt) af[mt] = loadA(cur, mt);
#pragma unroll
        for (int nt = 0; nt < 4; ++nt) bf[nt] = loadB(cur, nt);

#pragma unroll
        for (int mt = 0; mt < 2; ++mt)
#pragma unroll
            for (int nt = 0; nt < 4; ++nt)
                acc[mt][nt] = __builtin_amdgcn_wmma_f32_16x16x32_bf16(
                    /*neg_a=*/false, af[mt], /*neg_b=*/false, bf[nt],
                    /*c_mod=*/(short)0, acc[mt][nt],
                    /*reuse_a=*/false, /*reuse_b=*/false);
    }

    // Epilogue: out[row, oblk*BLK+col] = x + bias[k,col] + acc.
    // C/D layout: VGPR r, lane L -> (M = r + 8*(L>>4), N = L&15).
    const int oblk = (k + 1) & (KBLK - 1);
    const float* bp = bias + (long long)k * BLK;
    const int n   = lane & 15;
    const int mhi = (lane >> 4) * 8;
#pragma unroll
    for (int mt = 0; mt < 2; ++mt)
#pragma unroll
        for (int nt = 0; nt < 4; ++nt) {
            const int col = nt0 + waveN * 64 + nt * 16 + n;     // col within block
            const float bb = bp[col];
            const long long gcol = (long long)oblk * BLK + col;
#pragma unroll
            for (int r = 0; r < 8; ++r) {
                const int row = mt0 + waveM * 32 + mt * 16 + mhi + r;
                const long long idx = (long long)row * TOTAL + gcol;
                out[idx] = x[idx] + bb + acc[mt][nt][r];
            }
        }
}

// ---------------------------------------------------------------------------
// Launch: prep (tanh->bf16, W->bf16) into workspace, then 8x batched GEMM.
// Workspace use: H bf16 = 64MB @ 0, W bf16 = 16MB @ 64MB (80MB total).
// ---------------------------------------------------------------------------
extern "C" void kernel_launch(void* const* d_in, const int* in_sizes, int n_in,
                              void* d_out, int out_size, void* d_ws, size_t ws_size,
                              hipStream_t stream) {
    const float* x = (const float*)d_in[0];
    const float* v = (const float*)d_in[1];
    const float* W = (const float*)d_in[2];
    const float* b = (const float*)d_in[3];
    float* out = (float*)d_out;

    unsigned short* Hbf = (unsigned short*)d_ws;
    unsigned short* Wbf = (unsigned short*)((char*)d_ws + (size_t)BATCH * TOTAL * 2);

    const long long nH = (long long)BATCH * TOTAL;        // 33554432
    const long long nW = (long long)KBLK * BLK * BLK;     // 8388608

    tanh_to_bf16<<<(unsigned)(nH / (256 * 8)), 256, 0, stream>>>(v, Hbf);
    conv_to_bf16<<<(unsigned)(nW / (256 * 8)), 256, 0, stream>>>(W, Wbf);

    dim3 grid(BLK / BN, BATCH / BM, KBLK);                // (8, 32, 8) = 2048 WGs
    block_gemm<<<grid, 256, 0, stream>>>(Hbf, Wbf, x, b, out);
}
